// GraphMatHighwayLayers_23914377904732
// MI455X (gfx1250) — matrix operand order repaired and checked by reading, compiled, not run on hardware
//
#include <hip/hip_runtime.h>

typedef __attribute__((ext_vector_type(16))) _Float16 v16h;
typedef __attribute__((ext_vector_type(8)))  _Float16 v8h;
typedef __attribute__((ext_vector_type(8)))  float    v8f;
typedef __attribute__((ext_vector_type(4)))  unsigned int u32x4;
typedef __attribute__((ext_vector_type(8)))  unsigned int u32x8;

#define GSZ 4
#define MM  64
#define CC  128
#define LL  4

// LDS layout (bytes):
//   h32  [64][128]  f32 :      0 ..  32768
//   h16  [64][128]  f16 :  32768 ..  49152
//   G16  [4][64][64]f16 :  49152 ..  81920
//   WT0  [128][128] f16 :  81920 .. 114688   (double-buffered W^T, TDM dest)
//   WT1  [128][128] f16 : 114688 .. 147456
//   mxT  [128][64]  f16 : 147456 .. 163840   (mx transposed: mxT[p][m])
//   hw32 [64][128][4]f32: 163840 .. 294912   (highway accumulator, coalesced flush)
#define SMEM_BYTES 294912

// ---- TDM: issue a 2D f16 tensor_load_to_lds (rows x cols, row stride in elems) ----
__device__ __forceinline__ void tdm_load_2d_f16(unsigned lds_off, const void* gaddr,
                                                unsigned rows, unsigned cols,
                                                unsigned stride_elems) {
    union { u32x4 v; unsigned u[4]; } g0;
    union { u32x8 v; unsigned u[8]; } g1;
    unsigned long long ga = (unsigned long long)gaddr;
    g0.u[0] = 1u;                                              // count=1, user D#
    g0.u[1] = lds_off;                                         // lds_addr (bytes)
    g0.u[2] = (unsigned)(ga & 0xFFFFFFFFu);                    // global_addr[31:0]
    g0.u[3] = (unsigned)((ga >> 32) & 0x1FFFFFFu) | (2u << 30);// global_addr[56:32], type=2
    g1.u[0] = (1u << 16);                                      // mask=0, data_size=1 (2B)
    g1.u[1] = (cols & 0xFFFFu) << 16;                          // tensor_dim0[15:0] @ bit48
    g1.u[2] = ((cols >> 16) & 0xFFFFu) | ((rows & 0xFFFFu) << 16); // dim0 hi | dim1 lo
    g1.u[3] = ((rows >> 16) & 0xFFFFu) | ((cols & 0xFFFFu) << 16); // dim1 hi | tile_dim0
    g1.u[4] = rows & 0xFFFFu;                                  // tile_dim1 (tile_dim2=0)
    g1.u[5] = stride_elems;                                    // tensor_dim0_stride lo32
    g1.u[6] = 0u;                                              // stride0 hi | stride1 lo
    g1.u[7] = 0u;
    asm volatile("tensor_load_to_lds %0, %1" :: "s"(g0.v), "s"(g1.v) : "memory");
}

// ---- pre-pass: W [L][GS][C][C] f32 -> ws WT16 [L*GS][p][c] f16 (transposed) ----
extern "C" __global__ void __launch_bounds__(256)
wconvert_kernel(const float* __restrict__ W, _Float16* __restrict__ WT) {
    int idx = blockIdx.x * 256 + threadIdx.x;      // over L*GS*C*C = 262144
    int s = idx >> 14;                             // (l,g) slab
    int r = idx & 16383;
    int c = r >> 7;                                // contraction index
    int p = r & 127;                               // output index
    WT[(s << 14) + p * CC + c] = (_Float16)W[idx];
}

extern "C" __global__ void __launch_bounds__(256, 1)
graphmat_highway_kernel(const float* __restrict__ Gm,
                        const float* __restrict__ X,
                        const _Float16* __restrict__ wsWT,
                        const float* __restrict__ Bv,
                        float* __restrict__ out_h,
                        float* __restrict__ out_hw)
{
    extern __shared__ char smem_raw[];
    float*    h32  = (float*)(smem_raw);
    _Float16* h16  = (_Float16*)(smem_raw + 32768);
    _Float16* G16  = (_Float16*)(smem_raw + 49152);
    _Float16* WTb0 = (_Float16*)(smem_raw + 81920);
    _Float16* mxT  = (_Float16*)(smem_raw + 147456);
    float*    hw32 = (float*)(smem_raw + 163840);
    const unsigned WT_LDS_OFF = 81920u;

    const int n    = blockIdx.x;
    const int tid  = threadIdx.x;
    const int lane = tid & 31;
    const int wave = tid >> 5;
    const int hs   = (lane >> 4) & 1;   // 16-lane half of the wave
    const int ln   = lane & 15;

    // kick off TDM for stage 0 (overlaps the x/G staging below)
    if (wave == 0)
        tdm_load_2d_f16(WT_LDS_OFF, wsWT, CC, CC, CC);

    // ---- stage x[n] -> h32 + h16 ----
    {
        const float4* xs = (const float4*)(X + (size_t)n * MM * CC);
        for (int i = tid; i < MM * CC / 4; i += 256) {
            float4 v = xs[i];
            ((float4*)h32)[i] = v;
            int j = i * 4;
            h16[j + 0] = (_Float16)v.x; h16[j + 1] = (_Float16)v.y;
            h16[j + 2] = (_Float16)v.z; h16[j + 3] = (_Float16)v.w;
        }
    }
    // ---- stage G[n] -> G16 (converted once, reused by all layers) ----
    {
        const float4* gs = (const float4*)(Gm + (size_t)n * GSZ * MM * MM);
        for (int i = tid; i < GSZ * MM * MM / 4; i += 256) {
            float4 v = gs[i];
            int j = i * 4;
            G16[j + 0] = (_Float16)v.x; G16[j + 1] = (_Float16)v.y;
            G16[j + 2] = (_Float16)v.z; G16[j + 3] = (_Float16)v.w;
        }
    }

    // 32 output tiles (4 m-tiles x 8 p-tiles); wave owns tiles wave*4 .. wave*4+3
    v8f yacc[4];

    for (int l = 0; l < LL; ++l) {
        #pragma unroll
        for (int q = 0; q < 4; ++q)
            #pragma unroll
            for (int r = 0; r < 8; ++r) yacc[q][r] = 0.0f;

        for (int g = 0; g < GSZ; ++g) {
            const int s = l * GSZ + g;
            __syncthreads();   // all waves done reading WT buf (s+1)&1 and mxT

            // prefetch next W slab via TDM into the other buffer; drain current
            if (wave == 0) {
                if (s + 1 < LL * GSZ) {
                    tdm_load_2d_f16(WT_LDS_OFF + (unsigned)(((s + 1) & 1) * 32768),
                                    wsWT + (size_t)(s + 1) * CC * CC, CC, CC, CC);
                    __builtin_amdgcn_s_wait_tensorcnt(1);   // stage s complete
                } else {
                    __builtin_amdgcn_s_wait_tensorcnt(0);
                }
            }
            __syncthreads();   // WT buf s&1 visible to all waves
            const _Float16* WT = WTb0 + (s & 1) * (CC * CC);

            // ---- matmul1: mx = h @ W[l][g] + b[l][g]; store transposed f16 ----
            for (int q = 0; q < 4; ++q) {
                int t  = wave * 4 + q;
                int m0 = (t >> 3) * 16;
                int n0 = (t & 7) * 16;
                v8f acc = {};
                #pragma unroll
                for (int k0 = 0; k0 < CC; k0 += 32) {
                    union { v16h v; v8h h[2]; } a;
                    const _Float16* ar = h16 + (m0 + ln) * CC + k0 + hs * 16;
                    a.h[0] = *(const v8h*)(ar);        // K 0..7  (hi half: 16..23)
                    a.h[1] = *(const v8h*)(ar + 8);    // K 8..15 (hi half: 24..31)
                    v16h bf = *(const v16h*)(WT + (n0 + ln) * CC + k0 + hs * 16);
                    acc = __builtin_amdgcn_wmma_f32_16x16x32_f16(
                        false, a.v, false, bf, (short)0, acc, false, false);
                }
                float bias = Bv[(l * GSZ + g) * CC + n0 + ln];
                int mrow = m0 + hs * 8;
                v8h st;
                #pragma unroll
                for (int r = 0; r < 8; ++r) st[r] = (_Float16)(acc[r] + bias);
                *(v8h*)(mxT + (n0 + ln) * MM + mrow) = st;
            }
            __syncthreads();

            // ---- matmul2: xo = G16[g] @ mx; accumulate relu(xo) in VGPRs ----
            for (int q = 0; q < 4; ++q) {
                int t  = wave * 4 + q;
                int m0 = (t >> 3) * 16;
                int n0 = (t & 7) * 16;
                v8f xo = {};
                #pragma unroll
                for (int k0 = 0; k0 < MM; k0 += 32) {
                    union { v16h v; v8h h[2]; } a;
                    const _Float16* ar = G16 + g * MM * MM + (m0 + ln) * MM + k0 + hs * 16;
                    a.h[0] = *(const v8h*)(ar);
                    a.h[1] = *(const v8h*)(ar + 8);
                    v16h bf = *(const v16h*)(mxT + (n0 + ln) * MM + k0 + hs * 16);
                    xo = __builtin_amdgcn_wmma_f32_16x16x32_f16(
                        false, a.v, false, bf, (short)0, xo, false, false);
                }
                #pragma unroll
                for (int r = 0; r < 8; ++r)
                    yacc[q][r] += fmaxf(xo[r], 0.0f);
            }
        }
        __syncthreads();

        // ---- residual update + highway into LDS (each (m,p) owned by one wave) ----
        for (int q = 0; q < 4; ++q) {
            int t  = wave * 4 + q;
            int mb = (t >> 3) * 16 + hs * 8;
            int p  = (t & 7) * 16 + ln;
            #pragma unroll
            for (int r = 0; r < 8; ++r) {
                int m = mb + r;
                float y  = yacc[q][r];
                float hn = y + h32[m * CC + p];
                h32[m * CC + p] = hn;
                h16[m * CC + p] = (_Float16)hn;
                hw32[(m * CC + p) * LL + l] = y;
            }
        }
    }
    __syncthreads();

    // ---- coalesced flush: final h and the full highway block ----
    {
        const float4* hv = (const float4*)h32;
        float4* od = (float4*)(out_h + (size_t)n * MM * CC);
        for (int i = tid; i < MM * CC / 4; i += 256) od[i] = hv[i];

        const float4* wv = (const float4*)hw32;
        float4* ow = (float4*)(out_hw + (size_t)n * MM * CC * LL);
        for (int i = tid; i < MM * CC * LL / 4; i += 256) ow[i] = wv[i];
    }
}

extern "C" void kernel_launch(void* const* d_in, const int* in_sizes, int n_in,
                              void* d_out, int out_size, void* d_ws, size_t ws_size,
                              hipStream_t stream) {
    (void)n_in; (void)out_size; (void)ws_size;
    const float* Gm = (const float*)d_in[0];   // [N, GS, M, M]
    const float* X  = (const float*)d_in[1];   // [N, M, C]
    const float* W  = (const float*)d_in[2];   // [L, GS, C, C]
    const float* Bv = (const float*)d_in[3];   // [L, GS, C]

    const int nGraphs = in_sizes[1] / (MM * CC);   // 1024
    float* out_h  = (float*)d_out;                               // [N, M, C]
    float* out_hw = (float*)d_out + (size_t)nGraphs * MM * CC;   // [N, M, C, L]

    _Float16* wsWT = (_Float16*)d_ws;          // [L*GS][C][C] f16, 512 KB

    // pre-convert W to f16-transposed slabs (L2-resident, shared by all blocks)
    wconvert_kernel<<<(LL * GSZ * CC * CC) / 256, 256, 0, stream>>>(W, wsWT);

    hipFuncSetAttribute((const void*)graphmat_highway_kernel,
                        hipFuncAttributeMaxDynamicSharedMemorySize, SMEM_BYTES);

    graphmat_highway_kernel<<<nGraphs, 256, SMEM_BYTES, stream>>>(
        Gm, X, wsWT, Bv, out_h, out_hw);
}
